// CausalSelfAttention_37933151158714
// MI455X (gfx1250) — matrix operand order repaired and checked by reading, not compiled, not run on hardware
//
#include <hip/hip_runtime.h>
#include <cstdint>

typedef __attribute__((ext_vector_type(16))) __bf16 v16bf;
typedef __attribute__((ext_vector_type(8)))  __bf16 v8bf;
typedef __attribute__((ext_vector_type(8)))  float  v8f;
typedef __attribute__((ext_vector_type(4)))  unsigned int u32x4;
typedef __attribute__((ext_vector_type(8)))  int i32x8;
typedef __attribute__((ext_vector_type(4)))  int i32x4;

#define B_ 4
#define T_ 2048
#define C_ 1024
#define H_ 16
#define D_ 64

// ---- helpers -------------------------------------------------------------

__device__ __forceinline__ unsigned short f2bf(float f) {
  unsigned int u = __builtin_bit_cast(unsigned int, f);
  u = (u + 0x7FFFu + ((u >> 16) & 1u)) >> 16;   // round-to-nearest-even
  return (unsigned short)u;
}

__device__ __forceinline__ v8f wmma_bf16(v16bf a, v16bf b, v8f c) {
  return __builtin_amdgcn_wmma_f32_16x16x32_bf16(false, a, false, b, (short)0, c,
                                                 false, false);
}

__device__ __forceinline__ v16bf join8(v8bf lo, v8bf hi) {
  return __builtin_shufflevector(lo, hi, 0, 1, 2, 3, 4, 5, 6, 7, 8, 9, 10, 11,
                                 12, 13, 14, 15);
}

// A fragment (16x32 bf16) from row-major [rows][ld]; lane m = row, contiguous
// K = 16*half + s (matches B fragments) -> two 16B loads.
__device__ __forceinline__ v16bf load_a(const unsigned short* base, int ld,
                                        int m, int k0, int half) {
  const v8bf lo = *(const v8bf*)(base + (size_t)m * ld + k0 + half * 16);
  const v8bf hi = *(const v8bf*)(base + (size_t)m * ld + k0 + half * 16 + 8);
  return join8(lo, hi);
}

// ---- Tensor Data Mover: 2-D tile DMA global -> LDS -----------------------
__device__ __forceinline__ void tdm_load_2d(unsigned lds_addr, const void* gptr,
                                            unsigned tile0, unsigned tile1,
                                            unsigned tdim0, unsigned tdim1,
                                            unsigned stride0) {
  const unsigned long long ga = (unsigned long long)gptr;
  u32x4 g0;
  g0[0] = 1u;                                                 // count=1 (valid)
  g0[1] = lds_addr;                                           // lds_addr bytes
  g0[2] = (unsigned)ga;                                       // global[31:0]
  g0[3] = (unsigned)((ga >> 32) & 0x01FFFFFFu) | (2u << 30);  // [56:32], type=2
  i32x8 g1;
  g1[0] = (int)(1u << 16);                        // data_size=1 -> 2B, mask=0
  g1[1] = (int)((tdim0 & 0xFFFFu) << 16);         // tensor_dim0[15:0]
  g1[2] = (int)((tdim0 >> 16) | ((tdim1 & 0xFFFFu) << 16));
  g1[3] = (int)((tdim1 >> 16) | (tile0 << 16));   // tile_dim0
  g1[4] = (int)(tile1 & 0xFFFFu);                 // tile_dim1 (tile_dim2=0)
  g1[5] = (int)stride0;                           // tensor_dim0_stride[31:0]
  g1[6] = 0;
  g1[7] = 0;
  const i32x4 z4 = {0, 0, 0, 0};
#if __clang_major__ >= 23
  const i32x8 z8 = {0, 0, 0, 0, 0, 0, 0, 0};
  __builtin_amdgcn_tensor_load_to_lds(g0, g1, z4, z4, z8, 0);
#else
  __builtin_amdgcn_tensor_load_to_lds(g0, g1, z4, z4, 0);
#endif
}

__device__ __forceinline__ unsigned lds_off(const void* p) {
  return (unsigned)(unsigned long long)p;  // generic LDS addr truncates to offset
}

// ---- precision conversion kernels ---------------------------------------

__global__ void __launch_bounds__(256) k_cvt_bf16(const float* __restrict__ in,
                                                  unsigned short* __restrict__ out,
                                                  int n) {
  int i = blockIdx.x * blockDim.x + threadIdx.x;
  if (i < n) out[i] = f2bf(in[i]);
}

// Wt[n*K + k] = bf16(W[k*N + n])
__global__ void __launch_bounds__(256) k_transpose_bf16(const float* __restrict__ W,
                                                        unsigned short* __restrict__ Wt,
                                                        int K, int N) {
  int i = blockIdx.x * blockDim.x + threadIdx.x;
  if (i < K * N) {
    int n = i / K, k = i - n * K;
    Wt[i] = f2bf(W[(size_t)k * N + n]);
  }
}

// ---- QKV GEMM with TDM-staged weights ------------------------------------
// Block: 8 waves x (32 rows x 64 cols) = 256 rows; all waves share one
// 64-col weight slab (64x64 bf16) DMA'd to LDS, double buffered.
// Each LDS B-fragment feeds two WMMAs (two A row groups): DS:WMMA = 1:1.

__global__ void __launch_bounds__(256) k_gemm_qkv(const unsigned short* __restrict__ Xbf,
                                                  const unsigned short* __restrict__ WaT,
                                                  const float* __restrict__ bias,
                                                  unsigned short* __restrict__ Qo,
                                                  unsigned short* __restrict__ Ko,
                                                  unsigned short* __restrict__ Vt) {
  const int lane = threadIdx.x & 31;
  const int wave = threadIdx.x >> 5;
  const int half = lane >> 4;
  const int m    = lane & 15;
  const int rbase = blockIdx.y * 256 + wave * 32;
  const int nbase = blockIdx.x * 64;
  const bool leader = (wave == 0);

  __shared__ __align__(32) unsigned short bsm[2][64 * 64];  // 2 x 8 KB

  const unsigned short* Bsrc = WaT + (size_t)nbase * C_;
  if (leader) tdm_load_2d(lds_off(&bsm[0][0]), Bsrc, 64, 64, C_, 64, C_);

  v8f acc[2][4] = {};
  const unsigned short* Arow0 = Xbf + (size_t)rbase * C_;
  const unsigned short* Arow1 = Arow0 + (size_t)16 * C_;
  for (int k0 = 0, it = 0; k0 < C_; k0 += 64, ++it) {
    if (leader) __builtin_amdgcn_s_wait_tensorcnt(0);
    __syncthreads();
    if (leader && (k0 + 64 < C_))
      tdm_load_2d(lds_off(&bsm[(it + 1) & 1][0]), Bsrc + k0 + 64, 64, 64, C_, 64, C_);
    const unsigned short* bs = &bsm[it & 1][0];
#pragma unroll
    for (int kc = 0; kc < 64; kc += 32) {
      // hoist all fragment loads so DS loads clause and share one wait
      const v16bf a0 = load_a(Arow0, C_, m, k0 + kc, half);
      const v16bf a1 = load_a(Arow1, C_, m, k0 + kc, half);
      v16bf bfrag[4];
#pragma unroll
      for (int t = 0; t < 4; ++t)
        bfrag[t] = *(const v16bf*)(bs + (size_t)(t * 16 + m) * 64 + kc + half * 16);
#pragma unroll
      for (int t = 0; t < 4; ++t) {
        acc[0][t] = wmma_bf16(a0, bfrag[t], acc[0][t]);
        acc[1][t] = wmma_bf16(a1, bfrag[t], acc[1][t]);
      }
    }
  }

  const int b  = rbase >> 11;
  const int t0 = rbase & (T_ - 1);
#pragma unroll
  for (int t = 0; t < 4; ++t) {
    const int n    = nbase + t * 16 + m;
    const float bv = bias[n];
    const int part = n >> 10;
    const int cc   = n & (C_ - 1);
    const int hh   = cc >> 6;
    const int dd   = cc & (D_ - 1);
    const size_t bh = (size_t)(b * H_ + hh);
#pragma unroll
    for (int g = 0; g < 2; ++g) {
#pragma unroll
      for (int j = 0; j < 8; ++j) {
        const int tt = t0 + g * 16 + j + 8 * half;
        const unsigned short val = f2bf(acc[g][t][j] + bv);
        if (part == 0)      Qo[(bh * T_ + tt) * D_ + dd] = val;
        else if (part == 1) Ko[(bh * T_ + tt) * D_ + dd] = val;
        else                Vt[(bh * D_ + dd) * T_ + tt] = val;
      }
    }
  }
}

// ---- flash attention with TDM-staged K/V tiles ---------------------------
// Block-uniform key loop so all 8 waves share each 32-key K/V tile; waves
// past their causal frontier see fully-masked tiles (corr=1, p=0: no-op).

__global__ void __launch_bounds__(256) k_flash(const unsigned short* __restrict__ Qi,
                                               const unsigned short* __restrict__ Ki,
                                               const unsigned short* __restrict__ Vti,
                                               unsigned short* __restrict__ Ybf) {
  const int lane = threadIdx.x & 31;
  const int wave = threadIdx.x >> 5;
  const int half = lane >> 4;
  const int m    = lane & 15;
  const int bh   = blockIdx.y;               // b*H + h
  const int h    = bh & (H_ - 1);
  const int qblock = blockIdx.x * 128;
  const int qw   = qblock + wave * 16;
  const bool leader = (wave == 0);

  const unsigned short* Qb = Qi  + (size_t)bh * T_ * D_;
  const unsigned short* Kb = Ki  + (size_t)bh * T_ * D_;
  const unsigned short* Vb = Vti + (size_t)bh * D_ * T_;

  const v16bf qa0 = load_a(Qb + (size_t)qw * D_, D_, m, 0, half);
  const v16bf qa1 = load_a(Qb + (size_t)qw * D_, D_, m, 32, half);

  v8f o0 = {}, o1 = {}, o2 = {}, o3 = {};
  float mrow[8], lrow[8];
#pragma unroll
  for (int j = 0; j < 8; ++j) { mrow[j] = -__builtin_inff(); lrow[j] = 0.f; }

  __shared__ __align__(32) unsigned short Ksm[2][32 * 64];   // keys x d
  __shared__ __align__(32) unsigned short Vsm[2][64 * 32];   // d x keys
  __shared__ __align__(32) unsigned short plds[8 * 16 * 32]; // P tile per wave
  unsigned short* pl = plds + wave * (16 * 32);

  const float scale = 0.125f;  // 1/sqrt(D)
  const int kkend = qblock + 128;

  if (leader) {
    tdm_load_2d(lds_off(&Ksm[0][0]), Kb, 64, 32, 64, T_, 64);
    tdm_load_2d(lds_off(&Vsm[0][0]), Vb, 32, 64, T_, 64, T_);
  }

  for (int kk = 0, it = 0; kk < kkend; kk += 32, ++it) {
    if (leader) __builtin_amdgcn_s_wait_tensorcnt(0);
    __syncthreads();
    if (leader && (kk + 32 < kkend)) {
      const int nb = (it + 1) & 1;
      tdm_load_2d(lds_off(&Ksm[nb][0]), Kb + (size_t)(kk + 32) * D_, 64, 32, 64, T_, 64);
      tdm_load_2d(lds_off(&Vsm[nb][0]), Vb + (kk + 32), 32, 64, T_, 64, T_);
    }
    const unsigned short* Ks = &Ksm[it & 1][0];
    const unsigned short* Vs = &Vsm[it & 1][0];

    // S tile 16q x 32kk from LDS-staged K
    v8f s0 = {}, s1 = {};
    {
      const unsigned short* kr0 = Ks + (size_t)m * 64;
      const unsigned short* kr1 = Ks + (size_t)(16 + m) * 64;
      const v16bf kb00 = *(const v16bf*)(kr0 + half * 16);
      const v16bf kb01 = *(const v16bf*)(kr0 + 32 + half * 16);
      const v16bf kb10 = *(const v16bf*)(kr1 + half * 16);
      const v16bf kb11 = *(const v16bf*)(kr1 + 32 + half * 16);
      s0 = wmma_bf16(qa0, kb00, s0);
      s0 = wmma_bf16(qa1, kb01, s0);
      s1 = wmma_bf16(qa0, kb10, s1);
      s1 = wmma_bf16(qa1, kb11, s1);
    }
    float corr[8];
#pragma unroll
    for (int j = 0; j < 8; ++j) {
      const int q = qw + j + 8 * half;
      float a = (kk + m      <= q) ? s0[j] * scale : -__builtin_inff();
      float b = (kk + 16 + m <= q) ? s1[j] * scale : -__builtin_inff();
      float t = fmaxf(a, b);
      t = fmaxf(t, __shfl_xor(t, 1));
      t = fmaxf(t, __shfl_xor(t, 2));
      t = fmaxf(t, __shfl_xor(t, 4));
      t = fmaxf(t, __shfl_xor(t, 8));
      const float mn = fmaxf(mrow[j], t);
      corr[j] = __expf(mrow[j] - mn);
      mrow[j] = mn;
      const float p0 = __expf(a - mn);
      const float p1 = __expf(b - mn);
      float rs = p0 + p1;
      rs += __shfl_xor(rs, 1);
      rs += __shfl_xor(rs, 2);
      rs += __shfl_xor(rs, 4);
      rs += __shfl_xor(rs, 8);
      lrow[j] = lrow[j] * corr[j] + rs;
      const int row = j + 8 * half;
      pl[row * 32 + m]      = f2bf(p0);
      pl[row * 32 + 16 + m] = f2bf(p1);
      o0[j] *= corr[j]; o1[j] *= corr[j]; o2[j] *= corr[j]; o3[j] *= corr[j];
    }
    // C-layout -> A-fragment re-layout via per-wave LDS slab (in-order DS)
    const v16bf pa = join8(*(const v8bf*)(pl + m * 32 + half * 16),
                           *(const v8bf*)(pl + m * 32 + half * 16 + 8));
    // O += P * V from LDS-staged Vt
    const v16bf vb0 = *(const v16bf*)(Vs + (size_t)(0  + m) * 32 + half * 16);
    const v16bf vb1 = *(const v16bf*)(Vs + (size_t)(16 + m) * 32 + half * 16);
    const v16bf vb2 = *(const v16bf*)(Vs + (size_t)(32 + m) * 32 + half * 16);
    const v16bf vb3 = *(const v16bf*)(Vs + (size_t)(48 + m) * 32 + half * 16);
    o0 = wmma_bf16(pa, vb0, o0);
    o1 = wmma_bf16(pa, vb1, o1);
    o2 = wmma_bf16(pa, vb2, o2);
    o3 = wmma_bf16(pa, vb3, o3);
  }

#pragma unroll
  for (int j = 0; j < 8; ++j) {
    const float inv = 1.0f / lrow[j];
    const size_t r = (size_t)(bh >> 4) * T_ + qw + j + 8 * half;
    unsigned short* yp = Ybf + r * C_ + h * D_;
    yp[0 + m]  = f2bf(o0[j] * inv);
    yp[16 + m] = f2bf(o1[j] * inv);
    yp[32 + m] = f2bf(o2[j] * inv);
    yp[48 + m] = f2bf(o3[j] * inv);
  }
}

// ---- output projection with TDM-staged weights, fp32 out -----------------

__global__ void __launch_bounds__(256) k_gemm_proj(const unsigned short* __restrict__ Ybf,
                                                   const unsigned short* __restrict__ WpT,
                                                   const float* __restrict__ bias,
                                                   float* __restrict__ out) {
  const int lane = threadIdx.x & 31;
  const int wave = threadIdx.x >> 5;
  const int half = lane >> 4;
  const int m    = lane & 15;
  const int rbase = blockIdx.y * 256 + wave * 32;
  const int nbase = blockIdx.x * 64;
  const bool leader = (wave == 0);

  __shared__ __align__(32) unsigned short bsm[2][64 * 64];

  const unsigned short* Bsrc = WpT + (size_t)nbase * C_;
  if (leader) tdm_load_2d(lds_off(&bsm[0][0]), Bsrc, 64, 64, C_, 64, C_);

  v8f acc[2][4] = {};
  const unsigned short* Arow0 = Ybf + (size_t)rbase * C_;
  const unsigned short* Arow1 = Arow0 + (size_t)16 * C_;
  for (int k0 = 0, it = 0; k0 < C_; k0 += 64, ++it) {
    if (leader) __builtin_amdgcn_s_wait_tensorcnt(0);
    __syncthreads();
    if (leader && (k0 + 64 < C_))
      tdm_load_2d(lds_off(&bsm[(it + 1) & 1][0]), Bsrc + k0 + 64, 64, 64, C_, 64, C_);
    const unsigned short* bs = &bsm[it & 1][0];
#pragma unroll
    for (int kc = 0; kc < 64; kc += 32) {
      const v16bf a0 = load_a(Arow0, C_, m, k0 + kc, half);
      const v16bf a1 = load_a(Arow1, C_, m, k0 + kc, half);
      v16bf bfrag[4];
#pragma unroll
      for (int t = 0; t < 4; ++t)
        bfrag[t] = *(const v16bf*)(bs + (size_t)(t * 16 + m) * 64 + kc + half * 16);
#pragma unroll
      for (int t = 0; t < 4; ++t) {
        acc[0][t] = wmma_bf16(a0, bfrag[t], acc[0][t]);
        acc[1][t] = wmma_bf16(a1, bfrag[t], acc[1][t]);
      }
    }
  }
#pragma unroll
  for (int t = 0; t < 4; ++t) {
    const int n = nbase + t * 16 + m;
    const float bv = bias[n];
#pragma unroll
    for (int g = 0; g < 2; ++g) {
#pragma unroll
      for (int j = 0; j < 8; ++j) {
        out[(size_t)(rbase + g * 16 + j + 8 * half) * C_ + n] = acc[g][t][j] + bv;
      }
    }
  }
}

// ---- host-side launch ----------------------------------------------------

extern "C" void kernel_launch(void* const* d_in, const int* in_sizes, int n_in,
                              void* d_out, int out_size, void* d_ws, size_t ws_size,
                              hipStream_t stream) {
  (void)in_sizes; (void)n_in; (void)out_size; (void)ws_size;
  const float* x      = (const float*)d_in[0];
  const float* W_attn = (const float*)d_in[1];
  const float* b_attn = (const float*)d_in[2];
  const float* W_proj = (const float*)d_in[3];
  const float* b_proj = (const float*)d_in[4];
  float* out = (float*)d_out;

  char* p = (char*)d_ws;
  auto take = [&](size_t bytes) {
    char* q = p;
    p += (bytes + 255) & ~(size_t)255;
    return q;
  };
  unsigned short* Xbf = (unsigned short*)take((size_t)B_ * T_ * C_ * 2);
  unsigned short* WaT = (unsigned short*)take((size_t)3 * C_ * C_ * 2);
  unsigned short* WpT = (unsigned short*)take((size_t)C_ * C_ * 2);
  unsigned short* Qb  = (unsigned short*)take((size_t)B_ * T_ * C_ * 2);
  unsigned short* Kb  = (unsigned short*)take((size_t)B_ * T_ * C_ * 2);
  unsigned short* Vb  = (unsigned short*)take((size_t)B_ * T_ * C_ * 2);
  unsigned short* Ybf = (unsigned short*)take((size_t)B_ * T_ * C_ * 2);

  {
    int n = B_ * T_ * C_;
    k_cvt_bf16<<<(n + 255) / 256, 256, 0, stream>>>(x, Xbf, n);
  }
  {
    int n = C_ * 3 * C_;
    k_transpose_bf16<<<(n + 255) / 256, 256, 0, stream>>>(W_attn, WaT, C_, 3 * C_);
  }
  {
    int n = C_ * C_;
    k_transpose_bf16<<<(n + 255) / 256, 256, 0, stream>>>(W_proj, WpT, C_, C_);
  }
  k_gemm_qkv<<<dim3(3 * C_ / 64, B_ * T_ / 256), 256, 0, stream>>>(Xbf, WaT, b_attn,
                                                                   Qb, Kb, Vb);
  k_flash<<<dim3(T_ / 128, B_ * H_), 256, 0, stream>>>(Qb, Kb, Vb, Ybf);
  k_gemm_proj<<<dim3(C_ / 64, B_ * T_ / 256), 256, 0, stream>>>(Ybf, WpT, b_proj, out);
}
